// SequenceSparseMoeBlock_78194174591487
// MI455X (gfx1250) — matrix operand order repaired and checked by reading, not compiled, not run on hardware
//
#include <hip/hip_runtime.h>
#include <math.h>

// Problem constants (match reference)
#define BB 8
#define TT 2048
#define HH 1024
#define FF 4096
#define EE 8

typedef __bf16 bf16_t;
typedef __attribute__((ext_vector_type(16))) bf16_t v16bf;
typedef __attribute__((ext_vector_type(8)))  float  v8f;

// ---------------------------------------------------------------------------
// CDNA5 async global->LDS copies (ASYNCcnt-tracked). The builtin takes
// int4-typed pointers (per hipcc diagnostic): (global src, lds dst, off, cpol)
// ---------------------------------------------------------------------------
#if __has_builtin(__builtin_amdgcn_global_load_async_to_lds_b128)
#define HAVE_ASYNC_LDS 1
typedef int v4i_ __attribute__((vector_size(16)));
typedef __attribute__((address_space(1))) v4i_ GlbV4;
typedef __attribute__((address_space(3))) v4i_ LdsV4;
__device__ __forceinline__ void async_cp16(unsigned short* lds_dst,
                                           const unsigned short* gsrc) {
    __builtin_amdgcn_global_load_async_to_lds_b128(
        (GlbV4*)gsrc, (LdsV4*)lds_dst, 0, 0);
}
__device__ __forceinline__ void wait_async0() {
#if __has_builtin(__builtin_amdgcn_s_wait_asynccnt)
    __builtin_amdgcn_s_wait_asynccnt(0);
#else
    asm volatile("s_wait_asynccnt 0x0" ::: "memory");
#endif
}
#else
#define HAVE_ASYNC_LDS 0
#endif

// ---------------------------------------------------------------------------
// f32 -> bf16 (round to nearest even), raw ushort
// ---------------------------------------------------------------------------
__device__ __forceinline__ unsigned short f32_to_bf16(float f) {
    unsigned int u = __float_as_uint(f);
    u += 0x7FFFu + ((u >> 16) & 1u);
    return (unsigned short)(u >> 16);
}

// ---------------------------------------------------------------------------
// WMMA fragment loads from LDS (bf16, 16x16x32, wave32 layouts)
// A 16x32: lanes 0-15 hold K[0..15];          lanes 16-31 hold K[16..31]
// B 32x16: lanes 0-15 hold K[0..15];          lanes 16-31 hold K[16..31]
// ---------------------------------------------------------------------------
__device__ __forceinline__ v16bf load_frag_a(const unsigned short* rowp, int lane) {
    int base = (lane >> 4) * 16;
    union { uint4 u[2]; v16bf v; } f;
    f.u[0] = *reinterpret_cast<const uint4*>(rowp + base);
    f.u[1] = *reinterpret_cast<const uint4*>(rowp + base + 8);
    return f.v;
}
__device__ __forceinline__ v16bf load_frag_b(const unsigned short* colp, int lane) {
    int base = (lane >> 4) * 16;
    union { uint4 u[2]; v16bf v; } f;
    f.u[0] = *reinterpret_cast<const uint4*>(colp + base);
    f.u[1] = *reinterpret_cast<const uint4*>(colp + base + 8);
    return f.v;
}

// ---------------------------------------------------------------------------
// Plain f32 -> bf16 conversion, 8 elements/thread (n divisible by 2048)
// ---------------------------------------------------------------------------
__global__ __launch_bounds__(256) void conv_kernel(
    const float* __restrict__ src, unsigned short* __restrict__ dst)
{
    size_t i = ((size_t)blockIdx.x * 256 + threadIdx.x) * 8;
    float4 a = *reinterpret_cast<const float4*>(&src[i]);
    float4 b = *reinterpret_cast<const float4*>(&src[i + 4]);
    union { unsigned short s[8]; uint4 u; } o;
    o.s[0] = f32_to_bf16(a.x); o.s[1] = f32_to_bf16(a.y);
    o.s[2] = f32_to_bf16(a.z); o.s[3] = f32_to_bf16(a.w);
    o.s[4] = f32_to_bf16(b.x); o.s[5] = f32_to_bf16(b.y);
    o.s[6] = f32_to_bf16(b.z); o.s[7] = f32_to_bf16(b.w);
    *reinterpret_cast<uint4*>(&dst[i]) = o.u;
}

// ---------------------------------------------------------------------------
// Transposing f32 [R][C] -> bf16 [C][R] converter (per-expert via blockIdx.z)
// 32x32 tiles through LDS, coalesced on both sides
// ---------------------------------------------------------------------------
__global__ __launch_bounds__(256) void transconv_kernel(
    const float* __restrict__ src, unsigned short* __restrict__ dst,
    int R, int C)
{
    const size_t eoff = (size_t)blockIdx.z * R * C;
    const int r0 = blockIdx.y * 32, c0 = blockIdx.x * 32;
    __shared__ unsigned short tbuf[32][36];
    const int tid = threadIdx.x;
    {
        const int row = tid >> 3, c4 = (tid & 7) * 4;
        float4 v = *reinterpret_cast<const float4*>(
            &src[eoff + (size_t)(r0 + row) * C + c0 + c4]);
        tbuf[c4 + 0][row] = f32_to_bf16(v.x);
        tbuf[c4 + 1][row] = f32_to_bf16(v.y);
        tbuf[c4 + 2][row] = f32_to_bf16(v.z);
        tbuf[c4 + 3][row] = f32_to_bf16(v.w);
    }
    __syncthreads();
    {
        const int c = tid >> 3, r4 = (tid & 7) * 4;
        *reinterpret_cast<uint2*>(&dst[eoff + (size_t)(c0 + c) * R + r0 + r4]) =
            *reinterpret_cast<const uint2*>(&tbuf[c][r4]);
    }
}

// ---------------------------------------------------------------------------
// Routing: per-sequence mean -> logits -> top-2 -> normalized combine weights
// ---------------------------------------------------------------------------
__global__ __launch_bounds__(256) void route_kernel(
    const float* __restrict__ X, const float* __restrict__ gw,
    int* __restrict__ top_i, float* __restrict__ top_w)
{
    const int b = blockIdx.x, tid = threadIdx.x;
    __shared__ float meanv[HH];
    __shared__ float red[256];
    __shared__ float logits[EE];

    const float* Xb = X + (size_t)b * TT * HH;
    for (int h = tid; h < HH; h += 256) {
        float s = 0.f;
        for (int t = 0; t < TT; ++t) s += Xb[(size_t)t * HH + h];
        meanv[h] = s * (1.0f / (float)TT);
    }
    __syncthreads();

    for (int e = 0; e < EE; ++e) {
        float p = 0.f;
        for (int h = tid; h < HH; h += 256) p += meanv[h] * gw[h * EE + e];
        red[tid] = p;
        __syncthreads();
        for (int s = 128; s > 0; s >>= 1) {
            if (tid < s) red[tid] += red[tid + s];
            __syncthreads();
        }
        if (tid == 0) logits[e] = red[0];
        __syncthreads();
    }

    if (tid == 0) {
        float m = logits[0];
        for (int e = 1; e < EE; ++e) m = fmaxf(m, logits[e]);
        float p[EE];
        for (int e = 0; e < EE; ++e) p[e] = __expf(logits[e] - m);
        int i0 = 0; float p0 = p[0];
        for (int e = 1; e < EE; ++e) if (p[e] > p0) { p0 = p[e]; i0 = e; }
        int i1 = (i0 == 0) ? 1 : 0; float p1 = p[i1];
        for (int e = 0; e < EE; ++e)
            if (e != i0 && p[e] > p1) { p1 = p[e]; i1 = e; }
        float s = p0 + p1;
        top_i[b * 2 + 0] = i0;  top_i[b * 2 + 1] = i1;
        top_w[b * 2 + 0] = p0 / s;  top_w[b * 2 + 1] = p1 / s;
    }
}

// ---------------------------------------------------------------------------
// Shared staging helper: copy 128x32 bf16 tile (rows stride `gstride`) to LDS
// r = tid>>1 selects row, (tid&1)*16 selects 16-element half-row
// ---------------------------------------------------------------------------
__device__ __forceinline__ void stage_bf16_tile(
    unsigned short (*lds)[40], const unsigned short* gbase, size_t gstride,
    int tid)
{
    const int r = tid >> 1, co = (tid & 1) * 16;
    const unsigned short* g = gbase + (size_t)r * gstride + co;
#if HAVE_ASYNC_LDS
    async_cp16(&lds[r][co], g);
    async_cp16(&lds[r][co + 8], g + 8);
#else
    *reinterpret_cast<uint4*>(&lds[r][co])     = *reinterpret_cast<const uint4*>(g);
    *reinterpret_cast<uint4*>(&lds[r][co + 8]) = *reinterpret_cast<const uint4*>(g + 8);
#endif
}

// ---------------------------------------------------------------------------
// GEMM1: hmid[slot] = combine_w * gelu(X[b] @ w1[e] + b1[e]), bf16 out
// grid (FF/128, TT/128, 2), block 256
// PRE=true: bf16 pre-converted operands (X row-major, w1T [F][H])
// ---------------------------------------------------------------------------
template <bool PRE>
__global__ __launch_bounds__(256) void gemm1_kernel(
    const float* __restrict__ Xf, const unsigned short* __restrict__ Xb16,
    const float* __restrict__ w1f, const unsigned short* __restrict__ w1T,
    const float* __restrict__ b1,
    const int* __restrict__ top_i, const float* __restrict__ top_w,
    unsigned short* __restrict__ hmid, int b)
{
    const int slot  = blockIdx.z;
    const int e     = top_i[b * 2 + slot];
    const float wsl = top_w[b * 2 + slot];
    const float* bias = b1 + (size_t)e * FF;
    unsigned short* out = hmid + (size_t)slot * TT * FF;

    const int n0 = blockIdx.x * 128;   // F tile
    const int m0 = blockIdx.y * 128;   // T tile

    __shared__ unsigned short As[128][40];  // [m][k]
    __shared__ unsigned short Bs[128][40];  // [n][k]

    const int tid  = threadIdx.x;
    const int lane = tid & 31;
    const int wave = tid >> 5;
    const int wrow = (wave >> 1) * 32;
    const int wcol = (wave & 1) * 64;

    v8f acc[2][4] = {};

    for (int k0 = 0; k0 < HH; k0 += 32) {
        if (PRE) {
            const unsigned short* Xg = Xb16 + (size_t)b * TT * HH + (size_t)m0 * HH + k0;
            const unsigned short* Wg = w1T  + (size_t)e * FF * HH + (size_t)n0 * HH + k0;
            __builtin_prefetch(Wg + 32, 0, 1);
            stage_bf16_tile(As, Xg, HH, tid);
            stage_bf16_tile(Bs, Wg, HH, tid);
#if HAVE_ASYNC_LDS
            wait_async0();
#endif
        } else {
            const float* Xb = Xf  + (size_t)b * TT * HH;
            const float* W  = w1f + (size_t)e * HH * FF;
            __builtin_prefetch(&W[(size_t)(k0 + 32) * FF + n0], 0, 1);
            {
                const int r = tid >> 3, c = (tid & 7) * 4;
                #pragma unroll
                for (int i = 0; i < 4; ++i) {
                    float4 v = *reinterpret_cast<const float4*>(
                        &Xb[(size_t)(m0 + r + 32 * i) * HH + k0 + c]);
                    As[r + 32 * i][c + 0] = f32_to_bf16(v.x);
                    As[r + 32 * i][c + 1] = f32_to_bf16(v.y);
                    As[r + 32 * i][c + 2] = f32_to_bf16(v.z);
                    As[r + 32 * i][c + 3] = f32_to_bf16(v.w);
                }
            }
            {
                const int kr = tid >> 5, c = (tid & 31) * 4;
                #pragma unroll
                for (int i = 0; i < 4; ++i) {
                    float4 v = *reinterpret_cast<const float4*>(
                        &W[(size_t)(k0 + kr + 8 * i) * FF + n0 + c]);
                    Bs[c + 0][kr + 8 * i] = f32_to_bf16(v.x);
                    Bs[c + 1][kr + 8 * i] = f32_to_bf16(v.y);
                    Bs[c + 2][kr + 8 * i] = f32_to_bf16(v.z);
                    Bs[c + 3][kr + 8 * i] = f32_to_bf16(v.w);
                }
            }
        }
        __syncthreads();

        v16bf afr[2], bfr[4];
        #pragma unroll
        for (int i = 0; i < 2; ++i)
            afr[i] = load_frag_a(&As[wrow + i * 16 + (lane & 15)][0], lane);
        #pragma unroll
        for (int j = 0; j < 4; ++j)
            bfr[j] = load_frag_b(&Bs[wcol + j * 16 + (lane & 15)][0], lane);

        #pragma unroll
        for (int i = 0; i < 2; ++i)
            #pragma unroll
            for (int j = 0; j < 4; ++j)
                acc[i][j] = __builtin_amdgcn_wmma_f32_16x16x32_bf16(
                    false, afr[i], false, bfr[j], (short)0, acc[i][j], false, false);
        __syncthreads();
    }

    #pragma unroll
    for (int j = 0; j < 4; ++j) {
        const int col = n0 + wcol + j * 16 + (lane & 15);
        const float bv = bias[col];
        #pragma unroll
        for (int i = 0; i < 2; ++i) {
            #pragma unroll
            for (int v = 0; v < 8; ++v) {
                const int row = m0 + wrow + i * 16 + (lane >> 4) * 8 + v;
                float x = acc[i][j][v] + bv;
                float g = 0.5f * x * (1.0f + erff(x * 0.70710678118654752f));
                out[(size_t)row * FF + col] = f32_to_bf16(g * wsl);
            }
        }
    }
}

// ---------------------------------------------------------------------------
// GEMM2: out[b] = sum_slot hmid[slot] @ w2[e_slot] + sum_slot w_s * b2[e_slot]
// grid (HH/128, TT/128), block 256
// PRE=true: w2T [H][F] bf16 pre-converted
// ---------------------------------------------------------------------------
template <bool PRE>
__global__ __launch_bounds__(256) void gemm2_kernel(
    const unsigned short* __restrict__ hmid,
    const float* __restrict__ w2f, const unsigned short* __restrict__ w2T,
    const float* __restrict__ b2,
    const int* __restrict__ top_i, const float* __restrict__ top_w,
    float* __restrict__ out, int b)
{
    const int n0 = blockIdx.x * 128;   // H tile
    const int m0 = blockIdx.y * 128;   // T tile

    __shared__ unsigned short As[128][40];
    __shared__ unsigned short Bs[128][40];

    const int tid  = threadIdx.x;
    const int lane = tid & 31;
    const int wave = tid >> 5;
    const int wrow = (wave >> 1) * 32;
    const int wcol = (wave & 1) * 64;

    v8f acc[2][4] = {};

    for (int slot = 0; slot < 2; ++slot) {
        const int e = top_i[b * 2 + slot];
        const unsigned short* A = hmid + (size_t)slot * TT * FF;

        for (int k0 = 0; k0 < FF; k0 += 32) {
            // A tile: always bf16 in workspace
            stage_bf16_tile(As, A + (size_t)m0 * FF + k0, FF, tid);
            if (PRE) {
                const unsigned short* Wg =
                    w2T + (size_t)e * HH * FF + (size_t)n0 * FF + k0;
                __builtin_prefetch(Wg + 32, 0, 1);
                stage_bf16_tile(Bs, Wg, FF, tid);
#if HAVE_ASYNC_LDS
                wait_async0();
#endif
            } else {
                const float* W = w2f + (size_t)e * FF * HH;
                __builtin_prefetch(&W[(size_t)(k0 + 32) * HH + n0], 0, 1);
                const int kr = tid >> 5, c = (tid & 31) * 4;
                #pragma unroll
                for (int i = 0; i < 4; ++i) {
                    float4 v = *reinterpret_cast<const float4*>(
                        &W[(size_t)(k0 + kr + 8 * i) * HH + n0 + c]);
                    Bs[c + 0][kr + 8 * i] = f32_to_bf16(v.x);
                    Bs[c + 1][kr + 8 * i] = f32_to_bf16(v.y);
                    Bs[c + 2][kr + 8 * i] = f32_to_bf16(v.z);
                    Bs[c + 3][kr + 8 * i] = f32_to_bf16(v.w);
                }
#if HAVE_ASYNC_LDS
                wait_async0();   // drain A-tile async copies
#endif
            }
            __syncthreads();

            v16bf afr[2], bfr[4];
            #pragma unroll
            for (int i = 0; i < 2; ++i)
                afr[i] = load_frag_a(&As[wrow + i * 16 + (lane & 15)][0], lane);
            #pragma unroll
            for (int j = 0; j < 4; ++j)
                bfr[j] = load_frag_b(&Bs[wcol + j * 16 + (lane & 15)][0], lane);

            #pragma unroll
            for (int i = 0; i < 2; ++i)
                #pragma unroll
                for (int j = 0; j < 4; ++j)
                    acc[i][j] = __builtin_amdgcn_wmma_f32_16x16x32_bf16(
                        false, afr[i], false, bfr[j], (short)0, acc[i][j], false, false);
            __syncthreads();
        }
    }

    const int e0 = top_i[b * 2 + 0], e1 = top_i[b * 2 + 1];
    const float w0 = top_w[b * 2 + 0], w1v = top_w[b * 2 + 1];
    #pragma unroll
    for (int j = 0; j < 4; ++j) {
        const int col = n0 + wcol + j * 16 + (lane & 15);
        const float bv = w0 * b2[e0 * HH + col] + w1v * b2[e1 * HH + col];
        #pragma unroll
        for (int i = 0; i < 2; ++i) {
            #pragma unroll
            for (int v = 0; v < 8; ++v) {
                const int row = m0 + wrow + i * 16 + (lane >> 4) * 8 + v;
                out[((size_t)b * TT + row) * HH + col] = acc[i][j][v] + bv;
            }
        }
    }
}

// ---------------------------------------------------------------------------
// Launch
// ---------------------------------------------------------------------------
extern "C" void kernel_launch(void* const* d_in, const int* in_sizes, int n_in,
                              void* d_out, int out_size, void* d_ws, size_t ws_size,
                              hipStream_t stream) {
    const float* X  = (const float*)d_in[0];
    const float* gw = (const float*)d_in[1];
    const float* w1 = (const float*)d_in[2];
    const float* b1 = (const float*)d_in[3];
    const float* w2 = (const float*)d_in[4];
    const float* b2 = (const float*)d_in[5];
    float* out = (float*)d_out;

    char* ws = (char*)d_ws;
    int*   top_i = (int*)ws;
    float* top_w = (float*)(ws + 64);

    const size_t szX  = (size_t)BB * TT * HH * 2;          // 32 MB
    const size_t szW  = (size_t)EE * HH * FF * 2;          // 64 MB each
    const size_t szHm = (size_t)2 * TT * FF * 2;           // 32 MB
    const size_t need_pre = 256 + szX + 2 * szW + szHm;

    const bool pre = (ws_size >= need_pre);

    unsigned short *Xb16 = nullptr, *w1T = nullptr, *w2T = nullptr, *hmid;
    if (pre) {
        size_t off = 256;
        Xb16 = (unsigned short*)(ws + off); off += szX;
        w1T  = (unsigned short*)(ws + off); off += szW;
        w2T  = (unsigned short*)(ws + off); off += szW;
        hmid = (unsigned short*)(ws + off);
    } else {
        hmid = (unsigned short*)(ws + 256);
    }

    route_kernel<<<BB, 256, 0, stream>>>(X, gw, top_i, top_w);

    if (pre) {
        conv_kernel<<<(BB * TT * HH) / 2048, 256, 0, stream>>>(X, Xb16);
        transconv_kernel<<<dim3(FF / 32, HH / 32, EE), 256, 0, stream>>>(
            w1, w1T, HH, FF);   // [H][F] f32 -> [F][H] bf16
        transconv_kernel<<<dim3(HH / 32, FF / 32, EE), 256, 0, stream>>>(
            w2, w2T, FF, HH);   // [F][H] f32 -> [H][F] bf16
    }

    for (int b = 0; b < BB; ++b) {
        if (pre) {
            gemm1_kernel<true><<<dim3(FF / 128, TT / 128, 2), 256, 0, stream>>>(
                X, Xb16, w1, w1T, b1, top_i, top_w, hmid, b);
            gemm2_kernel<true><<<dim3(HH / 128, TT / 128), 256, 0, stream>>>(
                hmid, w2, w2T, b2, top_i, top_w, out, b);
        } else {
            gemm1_kernel<false><<<dim3(FF / 128, TT / 128, 2), 256, 0, stream>>>(
                X, Xb16, w1, w1T, b1, top_i, top_w, hmid, b);
            gemm2_kernel<false><<<dim3(HH / 128, TT / 128), 256, 0, stream>>>(
                hmid, w2, w2T, b2, top_i, top_w, out, b);
        }
    }
}
